// IGNEncoder_28939489641150
// MI455X (gfx1250) — matrix operand order repaired and checked by reading, not compiled, not run on hardware
//
#include <hip/hip_runtime.h>
#include <hip/hip_bf16.h>

typedef __attribute__((ext_vector_type(16))) _Float16 v16h;
typedef __attribute__((ext_vector_type(8)))  _Float16 v8h;
typedef __attribute__((ext_vector_type(8)))  float    v8f;
typedef __attribute__((ext_vector_type(4)))  float    v4f;

#define NB   32
#define MSZ  128
#define DIN  16
#define HID  64
#define OUTC 64

// ---------------- workspace layout (float offsets) ----------------
static constexpr size_t RS0 = 0;
static constexpr size_t CS0 = RS0 + (size_t)NB*DIN*MSZ;
static constexpr size_t DG0 = CS0 + (size_t)NB*DIN*MSZ;
static constexpr size_t SD0 = DG0 + (size_t)NB*DIN*MSZ;
static constexpr size_t TT0 = SD0 + (size_t)NB*DIN;
static constexpr size_t FR1 = TT0 + (size_t)NB*DIN;
static constexpr size_t FC1 = FR1 + (size_t)NB*HID*MSZ;
static constexpr size_t FD1 = FC1 + (size_t)NB*HID*MSZ;
static constexpr size_t RS1 = FD1 + (size_t)NB*HID*MSZ;
static constexpr size_t CS1 = RS1 + (size_t)NB*HID*MSZ;
static constexpr size_t DG1 = CS1 + (size_t)NB*HID*MSZ;
static constexpr size_t SD1 = DG1 + (size_t)NB*HID*MSZ;
static constexpr size_t TT1 = SD1 + (size_t)NB*HID;
static constexpr size_t FR2 = TT1 + (size_t)NB*HID;
static constexpr size_t FC2 = FR2 + (size_t)NB*HID*MSZ;
static constexpr size_t FD2 = FC2 + (size_t)NB*HID*MSZ;
static constexpr size_t RS2 = FD2 + (size_t)NB*HID*MSZ;
static constexpr size_t CS2 = RS2 + (size_t)NB*HID*MSZ;
static constexpr size_t DG2 = CS2 + (size_t)NB*HID*MSZ;
static constexpr size_t SD2 = DG2 + (size_t)NB*HID*MSZ;
static constexpr size_t TT2 = SD2 + (size_t)NB*HID;
static constexpr size_t RELU3 = TT2 + (size_t)NB*HID;
static constexpr size_t BNALL = RELU3 + (size_t)NB*OUTC*MSZ; // bn1[128] bn2[128] bn3[128]
static constexpr size_t AB1 = BNALL + 384;
static constexpr size_t AB2 = AB1 + 128;
static constexpr size_t AB3 = AB2 + 128;
static constexpr size_t FIRSTO = AB3 + 128;   // 32 ints
static constexpr size_t AB16O = FIRSTO + 32;  // 128 halfs (64 floats of space)
static constexpr size_t FLOATEND = AB16O + 64;
static constexpr size_t Z1B = ((FLOATEND*4 + 255)/256)*256;                 // byte offset
static constexpr size_t ZELEMS = (size_t)NB*MSZ*MSZ*HID;                    // halfs
static constexpr size_t Z2B = Z1B + ZELEMS*2;

// ---------------- small utility kernels ----------------
__global__ void kzero(float* p, int n) {
  int t = blockIdx.x*blockDim.x + threadIdx.x;
  if (t < n) p[t] = 0.f;
}

// reductions of raw f32 input (channel-last [n,i,j,d], D=16)
__global__ void k_rowred_f32(const float* __restrict__ x, float* __restrict__ rs,
                             float* __restrict__ dg) {
  int t = blockIdx.x*blockDim.x + threadIdx.x;      // NB*MSZ*DIN
  int d = t & (DIN-1); int i = (t >> 4) & (MSZ-1); int n = t >> 11;
  const float* p = x + ((size_t)(n*MSZ + i)*MSZ)*DIN + d;
  float acc = 0.f;
  for (int j = 0; j < MSZ; ++j) acc += p[(size_t)j*DIN];
  rs[((size_t)n*DIN + d)*MSZ + i] = acc;
  dg[((size_t)n*DIN + d)*MSZ + i] = p[(size_t)i*DIN];
}

__global__ void k_colred_f32(const float* __restrict__ x, float* __restrict__ cs) {
  int t = blockIdx.x*blockDim.x + threadIdx.x;      // NB*MSZ*DIN
  int d = t & (DIN-1); int j = (t >> 4) & (MSZ-1); int n = t >> 11;
  const float* p = x + ((size_t)n*MSZ*MSZ + j)*DIN + d;
  float acc = 0.f;
  for (int i = 0; i < MSZ; ++i) acc += p[(size_t)i*MSZ*DIN];
  cs[((size_t)n*DIN + d)*MSZ + j] = acc;
}

// reductions of f16 activations (channel-last [n,i,j,s], S=64) + BN sum/sumsq
__global__ void k_rowred_f16(const _Float16* __restrict__ z, float* __restrict__ rs,
                             float* __restrict__ dg, float* __restrict__ bn) {
  int t = blockIdx.x*blockDim.x + threadIdx.x;      // NB*MSZ*HID
  int d = t & (HID-1); int i = (t >> 6) & (MSZ-1); int n = t >> 13;
  const _Float16* p = z + ((size_t)(n*MSZ + i)*MSZ)*HID + d;
  float s = 0.f, q = 0.f;
  for (int j = 0; j < MSZ; ++j) { float v = (float)p[(size_t)j*HID]; s += v; q += v*v; }
  rs[((size_t)n*HID + d)*MSZ + i] = s;
  dg[((size_t)n*HID + d)*MSZ + i] = (float)p[(size_t)i*HID];
  atomicAdd(&bn[d], s);
  atomicAdd(&bn[HID + d], q);
}

__global__ void k_colred_f16(const _Float16* __restrict__ z, float* __restrict__ cs) {
  int t = blockIdx.x*blockDim.x + threadIdx.x;      // NB*MSZ*HID
  int d = t & (HID-1); int j = (t >> 6) & (MSZ-1); int n = t >> 13;
  const _Float16* p = z + ((size_t)n*MSZ*MSZ + j)*HID + d;
  float acc = 0.f;
  for (int i = 0; i < MSZ; ++i) acc += (float)p[(size_t)i*MSZ*HID];
  cs[((size_t)n*HID + d)*MSZ + j] = acc;
}

__global__ void k_sums(const float* __restrict__ rs, const float* __restrict__ dg,
                       float* __restrict__ sd, float* __restrict__ tt, int Din) {
  int t = blockIdx.x*blockDim.x + threadIdx.x;      // NB*Din
  if (t >= NB*Din) return;
  int n = t / Din, d = t % Din;
  float a = 0.f, b = 0.f;
  const float* pr = rs + ((size_t)n*Din + d)*MSZ;
  const float* pd = dg + ((size_t)n*Din + d)*MSZ;
  for (int i = 0; i < MSZ; ++i) { a += pd[i]; b += pr[i]; }
  sd[t] = a; tt[t] = b;
}

// row/col/diag feature vectors, stored channel-last [n, pos, s] for b128 epilogue loads
__global__ void k_feat(const float* __restrict__ rs, const float* __restrict__ cs,
                       const float* __restrict__ dg, const float* __restrict__ sd,
                       const float* __restrict__ tt, const float* __restrict__ coeffs,
                       int Din, const float* __restrict__ av, const float* __restrict__ bv,
                       const float* __restrict__ all_bias, const float* __restrict__ diag_bias,
                       float* __restrict__ FR, float* __restrict__ FC, float* __restrict__ FD) {
  int t = blockIdx.x*blockDim.x + threadIdx.x;      // NB*MSZ*HID
  int s = t & (HID-1); int i = (t >> 6) & (MSZ-1); int n = t >> 13;
  const float im = 1.f/128.f, im2 = 1.f/16384.f;
  float fr = 0.f, fc = 0.f, fd = 0.f, fk = 0.f, fkd = 0.f;
  for (int d = 0; d < Din; ++d) {
    float ad = av ? av[d] : 1.f;
    float bd = bv ? bv[d] : 0.f;
    size_t base = ((size_t)n*Din + d)*MSZ + i;
    float rsz = ad*rs[base] + 128.f*bd;
    float csz = ad*cs[base] + 128.f*bd;
    float dgz = ad*dg[base] + bd;
    float sdz = ad*sd[n*Din + d] + 128.f*bd;
    float ttz = ad*tt[n*Din + d] + 16384.f*bd;
    const float* c = coeffs + ((size_t)d*HID + s)*15;
    fr  += c[5]*csz*im + c[6]*rsz*im + c[11]*dgz;
    fc  += c[7]*csz*im + c[8]*rsz*im + c[12]*dgz;
    fd  += c[0]*dgz + c[2]*rsz*im + c[3]*csz*im;
    fk  += c[13]*sdz*im + c[14]*ttz*im2;
    fkd += c[1]*sdz*im + c[4]*ttz*im2;
  }
  size_t o = ((size_t)(n*MSZ + i))*HID + s;         // [n, pos, s]
  FR[o] = fr + fk + all_bias[s];
  FC[o] = fc;
  FD[o] = fd + fkd + diag_bias[s];
}

__global__ void k_abh(const float* __restrict__ ab, _Float16* __restrict__ abh) {
  int t = threadIdx.x;
  if (t < 128) abh[t] = (_Float16)ab[t];
}

__device__ __forceinline__ void load_f8(const float* p, float* f) {
  v4f a = *(const v4f*)p;
  v4f b = *(const v4f*)(p + 4);
#pragma unroll
  for (int r = 0; r < 4; ++r) { f[r] = a[r]; f[4 + r] = b[r]; }
}

// ---------------- WMMA layer 1: f32 input [n,i,j,16], K=32 ----------------
__global__ __launch_bounds__(256) void k_gemm1(const float* __restrict__ attn,
    const float* __restrict__ c1, const float* __restrict__ FR,
    const float* __restrict__ FC, const float* __restrict__ FD,
    _Float16* __restrict__ z1) {
  int bx = blockIdx.x;
  int n = bx >> 6; int tb = bx & 63;
  int I = (tb >> 3) << 4, J = (tb & 7) << 4;
  int lane = threadIdx.x & 31, wave = threadIdx.x >> 5;
  int col = lane & 15, hi = lane >> 4;

  // A operand (weights): 16x32, lane = s-row, k packed {16*hi..16*hi+15}
  v16h A[4];
  int kb = hi * 8;
#pragma unroll
  for (int st = 0; st < 4; ++st) {
    int s = st*16 + col;
#pragma unroll
    for (int h = 0; h < 16; ++h) {
      int k = hi*16 + h;
      float w = (k < 16) ? c1[((size_t)k*HID + s)*15 + 9]
                         : c1[((size_t)(k-16)*HID + s)*15 + 10];
      A[st][h] = (_Float16)w;
    }
  }

#pragma unroll
  for (int tpair = 0; tpair < 2; ++tpair) {
    int trow = wave*2 + tpair;
    int i = I + trow, j = J + col;
    // B operand: lanes hi=0 -> x[n,i,j,0..15]; hi=1 -> x[n,j,i,0..15] (k=16..31)
    const float* p = attn + ((hi == 0) ? ((size_t)(n*MSZ + i)*MSZ + j)
                                       : ((size_t)(n*MSZ + j)*MSZ + i)) * DIN;
    v4f q0 = *(const v4f*)(p);
    v4f q1 = *(const v4f*)(p + 4);
    v4f q2 = *(const v4f*)(p + 8);
    v4f q3 = *(const v4f*)(p + 12);
    v16h Bv;
#pragma unroll
    for (int h = 0; h < 4; ++h) {
      Bv[h]      = (_Float16)q0[h];
      Bv[h + 4]  = (_Float16)q1[h];
      Bv[h + 8]  = (_Float16)q2[h];
      Bv[h + 12] = (_Float16)q3[h];
    }
    float dsel = (i == j) ? 1.f : 0.f;
    const float* frp = FR + ((size_t)(n*MSZ + i))*HID;
    const float* fcp = FC + ((size_t)(n*MSZ + j))*HID;
    const float* fdp = FD + ((size_t)(n*MSZ + i))*HID;
#pragma unroll
    for (int st = 0; st < 4; ++st) {
      v8f c = {};
      c = __builtin_amdgcn_wmma_f32_16x16x32_f16(false, A[st], false, Bv,
                                                 (short)0, c, false, false);
      int sbase = st*16 + hi*8;
      float fr[8], fc[8], fd[8];
      load_f8(frp + sbase, fr);
      load_f8(fcp + sbase, fc);
      load_f8(fdp + sbase, fd);
      v8h outv;
#pragma unroll
      for (int r = 0; r < 8; ++r) {
        float v = c[r] + fr[r] + fc[r] + dsel*fd[r];
        outv[r] = (_Float16)fmaxf(v, 0.f);
      }
      *(v8h*)(z1 + ((size_t)(n*MSZ + i)*MSZ + j)*HID + sbase) = outv;
    }
  }
}

// ---------------- WMMA layer 2: f16 input [n,i,j,64] w/ BN affine, K=128 ----------------
__global__ __launch_bounds__(256) void k_gemm2(const _Float16* __restrict__ z1,
    const float* __restrict__ c2, const _Float16* __restrict__ ab16,
    const float* __restrict__ FR, const float* __restrict__ FC,
    const float* __restrict__ FD, _Float16* __restrict__ z2) {
  int bx = blockIdx.x;
  int n = bx >> 6; int tb = bx & 63;
  int I = (tb >> 3) << 4, J = (tb & 7) << 4;
  int lane = threadIdx.x & 31, wave = threadIdx.x >> 5;
  int col = lane & 15, hi = lane >> 4;

  v16h A[4][4];                       // [s-tile][k-step]
  int kb = hi * 8;
#pragma unroll
  for (int st = 0; st < 4; ++st) {
    int s = st*16 + col;
#pragma unroll
    for (int ks = 0; ks < 4; ++ks) {
#pragma unroll
      for (int h = 0; h < 16; ++h) {
        int k32 = hi*16 + h;
        int gk = ks*32 + k32;
        int op = (gk < 64) ? 9 : 10;
        int d  = gk & 63;
        A[st][ks][h] = (_Float16)c2[((size_t)d*HID + s)*15 + op];
      }
    }
  }

  // BN affine as f16, preloaded once: two possible 16-channel segments per lane
  int db0 = hi*16, db1 = 32 + hi*16;
  v8h a0l = *(const v8h*)(ab16 + db0);      v8h a0h = *(const v8h*)(ab16 + db0 + 8);
  v8h a1l = *(const v8h*)(ab16 + db1);      v8h a1h = *(const v8h*)(ab16 + db1 + 8);
  v8h b0l = *(const v8h*)(ab16 + 64 + db0); v8h b0h = *(const v8h*)(ab16 + 64 + db0 + 8);
  v8h b1l = *(const v8h*)(ab16 + 64 + db1); v8h b1h = *(const v8h*)(ab16 + 64 + db1 + 8);

#pragma unroll
  for (int tpair = 0; tpair < 2; ++tpair) {
    int trow = wave*2 + tpair;
    int i = I + trow, j = J + col;
    v8f acc[4] = {v8f{}, v8f{}, v8f{}, v8f{}};
#pragma unroll
    for (int ks = 0; ks < 4; ++ks) {
      int gkb = ks*32 + hi*16;                 // B: k-local = 16*hi + h
      int pi = (gkb < 64) ? i : j;
      int pj = (gkb < 64) ? j : i;
      int dbase = gkb & 63;
      const _Float16* p = z1 + ((size_t)(n*MSZ + pi)*MSZ + pj)*HID + dbase;
      v8h u0 = *(const v8h*)p;
      v8h u1 = *(const v8h*)(p + 8);
      v8h al = (ks & 1) ? a1l : a0l;  v8h ah = (ks & 1) ? a1h : a0h;
      v8h bl = (ks & 1) ? b1l : b0l;  v8h bh = (ks & 1) ? b1h : b0h;
      v16h Bv;
#pragma unroll
      for (int h = 0; h < 8; ++h) {
        Bv[h]     = al[h]*u0[h] + bl[h];
        Bv[h + 8] = ah[h]*u1[h] + bh[h];
      }
#pragma unroll
      for (int st = 0; st < 4; ++st)
        acc[st] = __builtin_amdgcn_wmma_f32_16x16x32_f16(false, A[st][ks], false, Bv,
                                                         (short)0, acc[st], false, false);
    }
    float dsel = (i == j) ? 1.f : 0.f;
    const float* frp = FR + ((size_t)(n*MSZ + i))*HID;
    const float* fcp = FC + ((size_t)(n*MSZ + j))*HID;
    const float* fdp = FD + ((size_t)(n*MSZ + i))*HID;
#pragma unroll
    for (int st = 0; st < 4; ++st) {
      int sbase = st*16 + hi*8;
      float fr[8], fc[8], fd[8];
      load_f8(frp + sbase, fr);
      load_f8(fcp + sbase, fc);
      load_f8(fdp + sbase, fd);
      v8h outv;
#pragma unroll
      for (int r = 0; r < 8; ++r) {
        float v = acc[st][r] + fr[r] + fc[r] + dsel*fd[r];
        outv[r] = (_Float16)fmaxf(v, 0.f);
      }
      *(v8h*)(z2 + ((size_t)(n*MSZ + i)*MSZ + j)*HID + sbase) = outv;
    }
  }
}

// ---------------- layer 3 (2->1) + relu + bn3 stats ----------------
__global__ void k_layer3(const float* __restrict__ rs, const float* __restrict__ cs,
                         const float* __restrict__ dg, const float* __restrict__ sd,
                         const float* __restrict__ tt, const float* __restrict__ av,
                         const float* __restrict__ bv, const float* __restrict__ c3,
                         const float* __restrict__ bias3, float* __restrict__ relu3,
                         float* __restrict__ bn3) {
  int t = blockIdx.x*blockDim.x + threadIdx.x;     // NB*OUTC*MSZ
  int i = t & (MSZ-1); int s = (t >> 7) & (OUTC-1); int n = t >> 13;
  float acc = bias3[s];
  for (int d = 0; d < HID; ++d) {
    float ad = av[d], bd = bv[d];
    size_t base = ((size_t)n*HID + d)*MSZ + i;
    float dgz = ad*dg[base] + bd;
    float rsz = ad*rs[base] + 128.f*bd;
    float csz = ad*cs[base] + 128.f*bd;
    float sdz = ad*sd[n*HID + d] + 128.f*bd;
    float ttz = ad*tt[n*HID + d] + 16384.f*bd;
    const float* c = c3 + ((size_t)d*OUTC + s)*5;
    acc += c[0]*dgz + (c[1]*sdz + c[2]*rsz + c[3]*csz)*(1.f/128.f) + c[4]*ttz*(1.f/16384.f);
  }
  acc = fmaxf(acc, 0.f);
  relu3[t] = acc;
  float ssum = acc, ssq = acc*acc;
#pragma unroll
  for (int m = 16; m >= 1; m >>= 1) {
    ssum += __shfl_xor(ssum, m, 32);
    ssq  += __shfl_xor(ssq,  m, 32);
  }
  if ((threadIdx.x & 31) == 0) {
    atomicAdd(&bn3[s], ssum);
    atomicAdd(&bn3[OUTC + s], ssq);
  }
}

__global__ void k_bnab(const float* __restrict__ bn, const float* __restrict__ gamma,
                       const float* __restrict__ beta, float cnt, float* __restrict__ ab) {
  int s = threadIdx.x;
  if (s < 64) {
    float mean = bn[s] / cnt;
    float var  = bn[64 + s] / cnt - mean*mean;
    float a = gamma[s] * rsqrtf(var + 1e-5f);
    ab[s] = a;
    ab[64 + s] = beta[s] - mean*a;
  }
}

__global__ void k_first(const int* __restrict__ batch, int* __restrict__ first, int ntot) {
  int t = blockIdx.x*blockDim.x + threadIdx.x;
  if (t < ntot) {
    if (t == 0 || batch[t] != batch[t-1]) first[batch[t]] = t;
  }
}

__global__ void k_out(const float* __restrict__ x, const int* __restrict__ batch,
                      const int* __restrict__ first, const float* __restrict__ relu3,
                      const float* __restrict__ ab3, float* __restrict__ out, int ntot) {
  int t = blockIdx.x*blockDim.x + threadIdx.x;
  if (t >= ntot*128) return;
  int row = t >> 7, c = t & 127;
  if (c < 64) {
    out[t] = x[(size_t)row*64 + c];
  } else {
    int s = c - 64;
    int g = batch[row];
    int pos = row - first[g];
    out[t] = ab3[s] * relu3[((size_t)g*OUTC + s)*MSZ + pos] + ab3[64 + s];
  }
}

// ---------------- launcher ----------------
extern "C" void kernel_launch(void* const* d_in, const int* in_sizes, int n_in,
                              void* d_out, int out_size, void* d_ws, size_t ws_size,
                              hipStream_t stream) {
  const float* attn  = (const float*)d_in[0];
  const float* xin   = (const float*)d_in[1];
  const int*   batch = (const int*)d_in[2];
  const float* c1    = (const float*)d_in[3];
  const float* dbias1= (const float*)d_in[4];
  const float* abias1= (const float*)d_in[5];
  const float* g1    = (const float*)d_in[6];
  const float* be1   = (const float*)d_in[7];
  const float* c2    = (const float*)d_in[8];
  const float* dbias2= (const float*)d_in[9];
  const float* abias2= (const float*)d_in[10];
  const float* g2    = (const float*)d_in[11];
  const float* be2   = (const float*)d_in[12];
  const float* c3    = (const float*)d_in[13];
  const float* bias3 = (const float*)d_in[14];
  const float* g3    = (const float*)d_in[15];
  const float* be3   = (const float*)d_in[16];

  float* W = (float*)d_ws;
  _Float16* z1 = (_Float16*)((char*)d_ws + Z1B);
  _Float16* z2 = (_Float16*)((char*)d_ws + Z2B);
  _Float16* ab16 = (_Float16*)(W + AB16O);
  int* firsti = (int*)(W + FIRSTO);
  int ntot = in_sizes[2];

  // zero BN accumulators
  kzero<<<1, 512, 0, stream>>>(W + BNALL, 384);

  // ---- layer 1 ----
  k_rowred_f32<<<(NB*MSZ*DIN)/256, 256, 0, stream>>>(attn, W+RS0, W+DG0);
  k_colred_f32<<<(NB*MSZ*DIN)/256, 256, 0, stream>>>(attn, W+CS0);
  k_sums<<<(NB*DIN + 63)/64, 64, 0, stream>>>(W+RS0, W+DG0, W+SD0, W+TT0, DIN);
  k_feat<<<(NB*HID*MSZ)/256, 256, 0, stream>>>(W+RS0, W+CS0, W+DG0, W+SD0, W+TT0,
      c1, DIN, nullptr, nullptr, abias1, dbias1, W+FR1, W+FC1, W+FD1);
  k_gemm1<<<NB*64, 256, 0, stream>>>(attn, c1, W+FR1, W+FC1, W+FD1, z1);

  // ---- bn1 stats + layer 2 features ----
  k_rowred_f16<<<(NB*MSZ*HID)/256, 256, 0, stream>>>(z1, W+RS1, W+DG1, W+BNALL);
  k_colred_f16<<<(NB*MSZ*HID)/256, 256, 0, stream>>>(z1, W+CS1);
  k_bnab<<<1, 64, 0, stream>>>(W+BNALL, g1, be1, 524288.f, W+AB1);
  k_abh<<<1, 128, 0, stream>>>(W+AB1, ab16);
  k_sums<<<(NB*HID + 63)/64, 64, 0, stream>>>(W+RS1, W+DG1, W+SD1, W+TT1, HID);
  k_feat<<<(NB*HID*MSZ)/256, 256, 0, stream>>>(W+RS1, W+CS1, W+DG1, W+SD1, W+TT1,
      c2, HID, W+AB1, W+AB1+64, abias2, dbias2, W+FR2, W+FC2, W+FD2);
  k_gemm2<<<NB*64, 256, 0, stream>>>(z1, c2, ab16, W+FR2, W+FC2, W+FD2, z2);

  // ---- bn2 stats + layer 3 ----
  k_rowred_f16<<<(NB*MSZ*HID)/256, 256, 0, stream>>>(z2, W+RS2, W+DG2, W+BNALL+128);
  k_colred_f16<<<(NB*MSZ*HID)/256, 256, 0, stream>>>(z2, W+CS2);
  k_bnab<<<1, 64, 0, stream>>>(W+BNALL+128, g2, be2, 524288.f, W+AB2);
  k_sums<<<(NB*HID + 63)/64, 64, 0, stream>>>(W+RS2, W+DG2, W+SD2, W+TT2, HID);
  k_layer3<<<(NB*OUTC*MSZ)/256, 256, 0, stream>>>(W+RS2, W+CS2, W+DG2, W+SD2, W+TT2,
      W+AB2, W+AB2+64, c3, bias3, W+RELU3, W+BNALL+256);
  k_bnab<<<1, 64, 0, stream>>>(W+BNALL+256, g3, be3, 4096.f, W+AB3);

  // ---- gather + concat ----
  k_first<<<(ntot + 255)/256, 256, 0, stream>>>(batch, firsti, ntot);
  k_out<<<(ntot*128 + 255)/256, 256, 0, stream>>>(xin, batch, firsti, W+RELU3, W+AB3,
      (float*)d_out, ntot);
}
